// CTCDecoderV1_15221364097064
// MI455X (gfx1250) — matrix operand order repaired and checked by reading, not compiled, not run on hardware
//
#include <hip/hip_runtime.h>
#include <math.h>

#define HIDN 512
#define NCLS 64
#define BB   32
#define SSEQ 1024

typedef __attribute__((ext_vector_type(16))) __bf16 v16bf;
typedef __attribute__((ext_vector_type(8)))  float  v8f;
typedef __attribute__((ext_vector_type(4)))  unsigned int v4u;

__device__ __forceinline__ unsigned short f2bf(float f) {
  unsigned int u = __float_as_uint(f);
  u += 0x7FFFu + ((u >> 16) & 1u);
  return (unsigned short)(u >> 16);
}

// ---------------- fp32 -> bf16 elementwise convert ----------------
__global__ __launch_bounds__(256) void k_cvt_bf16(const float* __restrict__ in,
                                                  unsigned short* __restrict__ out,
                                                  long long n) {
  long long i = (long long)blockIdx.x * blockDim.x + threadIdx.x;
  long long stride = (long long)gridDim.x * blockDim.x;
  for (; i < n; i += stride) out[i] = f2bf(in[i]);
}

// ---------------- W[K,N] fp32 -> Wt[N,K] bf16 ----------------
__global__ __launch_bounds__(256) void k_transpose_bf16(const float* __restrict__ W,
                                                        unsigned short* __restrict__ Wt,
                                                        int K, int N) {
  long long total = (long long)K * N;
  long long i = (long long)blockIdx.x * 256 + threadIdx.x;
  if (i >= total) return;
  int n = (int)(i / K);
  int k = (int)(i % K);
  Wt[(long long)n * K + k] = f2bf(W[(long long)k * N + n]);
}

// ---------------- batched bf16 WMMA GEMM ----------------
// C[M,N] = scale * (A[M,K] @ Bt[N,K]^T) + bias[N]
// OUT_MODE: 0 = fp32 row-major, 1 = bf16 row-major, 2 = bf16 transposed ([N][ldc])
// Block tile 128 x (32*SN), K-step 32, 256 threads = 8 waves (4 in M x 2 in N),
// wave tile 32 x (16*SN). Double-buffered LDS, distance-2 register prefetch.
template<int OUT_MODE, int SN>
__global__ __launch_bounds__(256)
void k_gemm_bf16(const unsigned short* __restrict__ A, long long strideAb,
                 const unsigned short* __restrict__ Bt, long long strideBb,
                 const float* __restrict__ bias,
                 void* __restrict__ C, long long strideCb,
                 int M, int N, int K, int ldc, float scale)
{
  constexpr int BN = 32 * SN;       // block tile N
  constexpr int NBCH = BN / 64;     // 16B chunks of B tile per thread (1 or 2)
  __shared__ __align__(16) unsigned short lA[2][128 * 32];
  __shared__ __align__(16) unsigned short lB[2][BN * 32];

  const int z = blockIdx.z;
  A  += (long long)z * strideAb;
  Bt += (long long)z * strideBb;

  const int m0 = blockIdx.y * 128;
  const int n0 = blockIdx.x * BN;
  const int t = threadIdx.x;
  const int wave = t >> 5, lane = t & 31;
  const int laneLo = lane & 15, laneHi = lane >> 4;
  const int waveM = wave >> 1, waveN = wave & 1;

  v4u regA[2], regB[NBCH];

  auto glbLoad = [&](int k0) {
    #pragma unroll
    for (int i = 0; i < 2; ++i) {
      int c = t + i * 256;
      int row = c >> 2, col = (c & 3) * 8;
      regA[i] = *(const v4u*)&A[(long long)(m0 + row) * K + k0 + col];
    }
    #pragma unroll
    for (int i = 0; i < NBCH; ++i) {
      int c = t + i * 256;
      int row = c >> 2, col = (c & 3) * 8;
      regB[i] = *(const v4u*)&Bt[(long long)(n0 + row) * K + k0 + col];
    }
  };
  auto dsStore = [&](int s) {
    #pragma unroll
    for (int i = 0; i < 2; ++i) {
      int c = t + i * 256;
      int row = c >> 2, col = (c & 3) * 8;
      *(v4u*)&lA[s][row * 32 + col] = regA[i];
    }
    #pragma unroll
    for (int i = 0; i < NBCH; ++i) {
      int c = t + i * 256;
      int row = c >> 2, col = (c & 3) * 8;
      *(v4u*)&lB[s][row * 32 + col] = regB[i];
    }
  };

  v8f acc[2][SN];
  #pragma unroll
  for (int i = 0; i < 2; ++i)
    #pragma unroll
    for (int j = 0; j < SN; ++j)
      #pragma unroll
      for (int r = 0; r < 8; ++r) acc[i][j][r] = 0.f;

  const int T = K >> 5;
  glbLoad(0);
  dsStore(0);
  if (T > 1) glbLoad(32);
  __syncthreads();

  for (int i = 0; i < T; ++i) {
    const int s = i & 1;
    // WMMA fragments per ISA 16-bit layouts:
    // A 16x32: lane(lo)=row M; halves 0..7 -> K=8*hi+h, 8..15 -> K=16+8*hi+(h-8)
    // B 32x16 (Bt rows): lane(lo)=col N; halves 0..7 -> K=8*hi+h, 8..15 -> K=16+8*hi+(h-8)
    v16bf afrag[2], bfrag[SN];
    #pragma unroll
    for (int sm = 0; sm < 2; ++sm) {
      int r = waveM * 32 + sm * 16 + laneLo;
      const unsigned short* p = &lA[s][r * 32 + laneHi * 8];
      union { v4u u[2]; v16bf v; } tmp;
      tmp.u[0] = *(const v4u*)p;
      tmp.u[1] = *(const v4u*)(p + 16);
      afrag[sm] = tmp.v;
    }
    #pragma unroll
    for (int sn = 0; sn < SN; ++sn) {
      int r = waveN * (16 * SN) + sn * 16 + laneLo;
      const unsigned short* p = &lB[s][r * 32 + laneHi * 8];
      union { v4u u[2]; v16bf v; } tmp;
      tmp.u[0] = *(const v4u*)p;
      tmp.u[1] = *(const v4u*)(p + 16);
      bfrag[sn] = tmp.v;
    }
    #pragma unroll
    for (int sm = 0; sm < 2; ++sm)
      #pragma unroll
      for (int sn = 0; sn < SN; ++sn)
        acc[sm][sn] = __builtin_amdgcn_wmma_f32_16x16x32_bf16(
            false, afrag[sm], false, bfrag[sn], (short)0, acc[sm][sn],
            false, false);

    if (i + 1 < T) {
      dsStore(s ^ 1);                       // safe: nobody reads buf[s^1] now
      if (i + 2 < T) glbLoad((i + 2) << 5); // prefetch distance 2
      __syncthreads();                      // publish buf[s^1]
    }
  }

  // Epilogue. C/D layout: lane holds col n = laneLo (+tile), rows r + 8*laneHi.
  #pragma unroll
  for (int sm = 0; sm < 2; ++sm) {
    #pragma unroll
    for (int sn = 0; sn < SN; ++sn) {
      int mb = m0 + waveM * 32 + sm * 16 + laneHi * 8;
      int n  = n0 + waveN * (16 * SN) + sn * 16 + laneLo;
      float bia = bias ? bias[n] : 0.f;
      v8f v = acc[sm][sn];
      if (OUT_MODE == 0) {
        float* o = (float*)C + (long long)z * strideCb;
        #pragma unroll
        for (int r = 0; r < 8; ++r)
          o[(long long)(mb + r) * ldc + n] = v[r] * scale + bia;
      } else if (OUT_MODE == 1) {
        unsigned short* o = (unsigned short*)C + (long long)z * strideCb;
        #pragma unroll
        for (int r = 0; r < 8; ++r)
          o[(long long)(mb + r) * ldc + n] = f2bf(v[r] * scale + bia);
      } else {
        // transposed bf16: out[n][m], lane's 8 rows contiguous -> one 16B store
        unsigned short* o = (unsigned short*)C + (long long)z * strideCb;
        union { unsigned short sh[8]; v4u u; } pk;
        #pragma unroll
        for (int r = 0; r < 8; ++r) pk.sh[r] = f2bf(v[r] * scale + bia);
        *(v4u*)&o[(long long)n * ldc + mb] = pk.u;
      }
    }
  }
}

// ---------------- masked softmax over attn rows, fp32 + bf16 outputs ---------
// one wave32 per row of length S=1024 (32 elems/lane)
__global__ __launch_bounds__(256)
void k_softmax_attn(float* __restrict__ attn, unsigned short* __restrict__ attn_bf,
                    const unsigned char* __restrict__ mask) {
  const int wave = threadIdx.x >> 5, lane = threadIdx.x & 31;
  const long long row = (long long)blockIdx.x * 8 + wave;   // 0 .. B*S-1
  const int b = (int)(row >> 10);                           // row / S
  float* p = attn + row * SSEQ;
  unsigned short* pb = attn_bf + row * SSEQ;
  const unsigned char* mrow = mask + (long long)b * SSEQ;
  float x[32];
  float mx = -3.0e38f;
  #pragma unroll
  for (int i = 0; i < 32; ++i) {
    int k = i * 32 + lane;
    float s = p[k];
    s = mrow[k] ? s : -1.0e9f;
    x[i] = s;
    mx = fmaxf(mx, s);
  }
  #pragma unroll
  for (int off = 16; off > 0; off >>= 1) mx = fmaxf(mx, __shfl_xor(mx, off, 32));
  float sum = 0.f;
  #pragma unroll
  for (int i = 0; i < 32; ++i) { x[i] = __expf(x[i] - mx); sum += x[i]; }
  #pragma unroll
  for (int off = 16; off > 0; off >>= 1) sum += __shfl_xor(sum, off, 32);
  float inv = 1.f / sum;
  #pragma unroll
  for (int i = 0; i < 32; ++i) {
    int k = i * 32 + lane;
    float pr = x[i] * inv;
    p[k] = pr;
    pb[k] = f2bf(pr);
  }
}

// ---------------- log_softmax over 64 classes, one wave32 per row ------------
__global__ __launch_bounds__(256)
void k_log_softmax(const float* __restrict__ logits, float* __restrict__ out) {
  const int wave = threadIdx.x >> 5, lane = threadIdx.x & 31;
  const long long row = (long long)blockIdx.x * 8 + wave;   // 0 .. B*S-1
  const float* p = logits + row * NCLS;
  float a0 = p[lane], a1 = p[lane + 32];
  float mx = fmaxf(a0, a1);
  #pragma unroll
  for (int off = 16; off > 0; off >>= 1) mx = fmaxf(mx, __shfl_xor(mx, off, 32));
  float sum = __expf(a0 - mx) + __expf(a1 - mx);
  #pragma unroll
  for (int off = 16; off > 0; off >>= 1) sum += __shfl_xor(sum, off, 32);
  float lse = mx + __logf(sum);
  float* o = out + row * NCLS;
  o[lane] = a0 - lse;
  o[lane + 32] = a1 - lse;
}

extern "C" void kernel_launch(void* const* d_in, const int* in_sizes, int n_in,
                              void* d_out, int out_size, void* d_ws, size_t ws_size,
                              hipStream_t stream) {
  (void)in_sizes; (void)n_in; (void)out_size; (void)ws_size;
  const float* hid          = (const float*)d_in[0];
  const unsigned char* mask = (const unsigned char*)d_in[1];
  const float* Wq = (const float*)d_in[2];
  const float* bq = (const float*)d_in[3];
  const float* Wk = (const float*)d_in[4];
  const float* bk = (const float*)d_in[5];
  const float* Wv = (const float*)d_in[6];
  const float* bv = (const float*)d_in[7];
  const float* W2 = (const float*)d_in[8];
  const float* b2 = (const float*)d_in[9];

  float* out_ls = (float*)d_out;                                   // [B,S,C]
  float* attn   = (float*)d_out + (long long)BB * SSEQ * NCLS;     // [B,S,S]

  char* ws = (char*)d_ws;
  const size_t MB = 1ull << 20;
  // Phase 1 regions
  unsigned short* hid_bf = (unsigned short*)(ws + 0);        // 32 MB
  unsigned short* Qb     = (unsigned short*)(ws + 32 * MB);  // 32 MB
  unsigned short* Kb     = (unsigned short*)(ws + 64 * MB);  // 32 MB
  unsigned short* Vt     = (unsigned short*)(ws + 96 * MB);  // 32 MB, [B][H][S]
  // Phase 2 regions (reuse: hid/Q dead after scores; K dead after scores; Vt dead after ctx)
  unsigned short* attnbf = (unsigned short*)(ws + 0);        // 64 MB, [B,S,S]
  unsigned short* ctxb   = (unsigned short*)(ws + 64 * MB);  // 32 MB, [B,S,H]
  float*          logits = (float*)(ws + 96 * MB);           // 8 MB
  // Weights (persistent)
  unsigned short* Wqt = (unsigned short*)(ws + 128 * MB);
  unsigned short* Wkt = Wqt + HIDN * HIDN;
  unsigned short* Wvt = Wkt + HIDN * HIDN;
  unsigned short* W2t = Wvt + HIDN * HIDN;

  const long long nHid = (long long)BB * SSEQ * HIDN;
  k_cvt_bf16<<<4096, 256, 0, stream>>>(hid, hid_bf, nHid);
  k_transpose_bf16<<<(HIDN * HIDN + 255) / 256, 256, 0, stream>>>(Wq, Wqt, HIDN, HIDN);
  k_transpose_bf16<<<(HIDN * HIDN + 255) / 256, 256, 0, stream>>>(Wk, Wkt, HIDN, HIDN);
  k_transpose_bf16<<<(HIDN * HIDN + 255) / 256, 256, 0, stream>>>(Wv, Wvt, HIDN, HIDN);
  k_transpose_bf16<<<(HIDN * NCLS + 255) / 256, 256, 0, stream>>>(W2, W2t, HIDN, NCLS);

  const float inv_sqrt_d = 0.04419417382415922f; // 1/sqrt(512)
  const long long sSH = (long long)SSEQ * HIDN;
  const long long sSS = (long long)SSEQ * SSEQ;

  // Q = hid @ Wq + bq   (bf16 row-major [B,S,H])
  k_gemm_bf16<1, 4><<<dim3(HIDN / 128, SSEQ / 128, BB), 256, 0, stream>>>(
      hid_bf, sSH, Wqt, 0, bq, Qb, sSH, SSEQ, HIDN, HIDN, HIDN, 1.f);
  // K = hid @ Wk + bk
  k_gemm_bf16<1, 4><<<dim3(HIDN / 128, SSEQ / 128, BB), 256, 0, stream>>>(
      hid_bf, sSH, Wkt, 0, bk, Kb, sSH, SSEQ, HIDN, HIDN, HIDN, 1.f);
  // Vt = (hid @ Wv + bv)^T  (bf16 [B,H,S])
  k_gemm_bf16<2, 4><<<dim3(HIDN / 128, SSEQ / 128, BB), 256, 0, stream>>>(
      hid_bf, sSH, Wvt, 0, bv, Vt, sSH, SSEQ, HIDN, HIDN, SSEQ, 1.f);
  // scores = Q @ K^T * 1/sqrt(d)  (fp32 into d_out attn region)
  k_gemm_bf16<0, 4><<<dim3(SSEQ / 128, SSEQ / 128, BB), 256, 0, stream>>>(
      Qb, sSH, Kb, sSH, nullptr, attn, sSS, SSEQ, SSEQ, HIDN, SSEQ, inv_sqrt_d);
  // masked softmax (fp32 in d_out, bf16 copy for next GEMM)
  k_softmax_attn<<<BB * SSEQ / 8, 256, 0, stream>>>(attn, attnbf, mask);
  // ctx = attn @ V  (bf16 row-major [B,S,H])
  k_gemm_bf16<1, 4><<<dim3(HIDN / 128, SSEQ / 128, BB), 256, 0, stream>>>(
      attnbf, sSS, Vt, sSH, nullptr, ctxb, sSH, SSEQ, HIDN, SSEQ, HIDN, 1.f);
  // logits = ctx @ W2 + b2  (fp32)
  k_gemm_bf16<0, 2><<<dim3(NCLS / 64, SSEQ / 128, BB), 256, 0, stream>>>(
      ctxb, sSH, W2t, 0, b2, logits, (long long)SSEQ * NCLS, SSEQ, NCLS, HIDN, NCLS, 1.f);
  // out = log_softmax(logits)
  k_log_softmax<<<BB * SSEQ / 8, 256, 0, stream>>>(logits, out_ls);
}
